// ContinuousSSMLayer_5987184411213
// MI455X (gfx1250) — matrix operand order repaired and checked by reading, not compiled, not run on hardware
//
#include <hip/hip_runtime.h>
#include <hip/hip_bf16.h>

// ---------------------------------------------------------------------------
// Continuous SSM (S5-style) layer for MI455X / gfx1250.
//   GEMM1: Bu(L,2P)  = u_f16(L,H) @ W1T(2P,H)^T        (v_wmma_f32_16x16x32_f16)
//   scan : chunked complex diagonal scan over L (f32)
//   GEMM2: out(L,H)  = X(L,2P) @ V2T(H,2P)^T + D*u     (v_wmma_f32_16x16x32_f16)
// ---------------------------------------------------------------------------

typedef __attribute__((ext_vector_type(16))) _Float16 v16h;
typedef __attribute__((ext_vector_type(8)))  _Float16 v8h;
typedef __attribute__((ext_vector_type(4)))  _Float16 v4h;
typedef __attribute__((ext_vector_type(8)))  float    v8f;

union V16 { v16h v; v8h h[2]; };

#define LL     16384   // sequence length
#define HH     1024    // io dim
#define NS     512     // 2*P (real-equivalent state width)
#define PP     256     // effective complex state dim
#define CHUNK  128     // scan chunk length
#define NCHUNK 128     // LL / CHUNK
#define BSCALE 256.0f  // pre-scale for B_bar -> f16 (undone in GEMM1 epilogue)

// ---------------------------------------------------------------------------
// K0: per-channel discretisation + lambda power table (1 block, 256 threads)
// ---------------------------------------------------------------------------
__global__ void setup_k(const float* __restrict__ Lre, const float* __restrict__ Lim,
                        const float* __restrict__ logstep,
                        float* __restrict__ lamBar, float* __restrict__ coef,
                        float* __restrict__ lamC, float* __restrict__ lampow) {
  int p = threadIdx.x;
  float dt = expf(logstep[p]);
  float lr = Lre[p], li = Lim[p];
  float er = expf(lr * dt);
  float ang = li * dt;
  float br = er * cosf(ang);          // lambda_bar (complex)
  float bi = er * sinf(ang);
  lamBar[2 * p] = br; lamBar[2 * p + 1] = bi;
  // coef = (lambda_bar - 1) / lambda = (lambda_bar - 1) * conj(lambda) / |lambda|^2
  float denom = lr * lr + li * li;
  float ar = br - 1.0f, ai = bi;
  coef[2 * p]     = (ar * lr + ai * li) / denom;
  coef[2 * p + 1] = (ai * lr - ar * li) / denom;
  // lampow[p][k] = lambda_bar^(k+1), k = 0..CHUNK-1 ; lamC = lambda_bar^CHUNK
  float xr = 1.0f, xi = 0.0f;
  for (int k = 0; k < CHUNK; ++k) {
    float nr = xr * br - xi * bi;
    float ni = xr * bi + xi * br;
    xr = nr; xi = ni;
    lampow[(p * CHUNK + k) * 2]     = xr;
    lampow[(p * CHUNK + k) * 2 + 1] = xi;
  }
  lamC[2 * p] = xr; lamC[2 * p + 1] = xi;
}

// ---------------------------------------------------------------------------
// K1: W1T (2P x H) f16, pre-transposed for WMMA-B, scaled by BSCALE
//     row n=p   : Re(coef_p * B_tilde[p,h]) ; row n=P+p : Im(...)
// ---------------------------------------------------------------------------
__global__ __launch_bounds__(256) void build_W1(const float* __restrict__ Bm,
                                                const float* __restrict__ coef,
                                                _Float16* __restrict__ W1T) {
  int q = blockIdx.x * 256 + threadIdx.x;       // PP*HH threads
  int h = q & (HH - 1);
  int p = q >> 10;
  float cr = coef[2 * p], ci = coef[2 * p + 1];
  size_t bidx = ((size_t)p * HH + h) * 2;
  float br = Bm[bidx], bi = Bm[bidx + 1];
  float re = cr * br - ci * bi;
  float im = cr * bi + ci * br;
  W1T[(size_t)p * HH + h]        = (_Float16)(re * BSCALE);
  W1T[(size_t)(PP + p) * HH + h] = (_Float16)(im * BSCALE);
}

// ---------------------------------------------------------------------------
// K2: V2T (H x 2P) f16, pre-transposed for WMMA-B of GEMM2.
//     V2T[h][p] = 2*C_re[h,p] ; V2T[h][P+p] = -2*C_im[h,p]   (conj_sym readout)
// ---------------------------------------------------------------------------
__global__ __launch_bounds__(256) void build_V2(const float* __restrict__ Cm,
                                                _Float16* __restrict__ V2T) {
  int q = blockIdx.x * 256 + threadIdx.x;       // HH*PP threads
  int p = q & (PP - 1);
  int h = q >> 8;
  size_t cidx = ((size_t)h * PP + p) * 2;
  V2T[(size_t)h * NS + p]      = (_Float16)( 2.0f * Cm[cidx]);
  V2T[(size_t)h * NS + PP + p] = (_Float16)(-2.0f * Cm[cidx + 1]);
}

// ---------------------------------------------------------------------------
// K3: u f32 -> f16 (4 elems/thread)
// ---------------------------------------------------------------------------
__global__ __launch_bounds__(256) void cvt_u(const float* __restrict__ U,
                                             _Float16* __restrict__ U16) {
  size_t q = ((size_t)blockIdx.x * 256 + threadIdx.x) * 4;
  float4 v = *(const float4*)(U + q);
  v4h o;
  o.x = (_Float16)v.x; o.y = (_Float16)v.y;
  o.z = (_Float16)v.z; o.w = (_Float16)v.w;
  *(v4h*)(U16 + q) = o;
}

// ---------------------------------------------------------------------------
// WMMA GEMM: Cm(M,N) = scale * A(M,K) @ BT(N,K)^T  [+ Dv[n]*U[m,n] if EPI]
// Block: 256 threads = 8 waves; block tile 128x128; wave tile 32x64 (2x4 frags)
// ---------------------------------------------------------------------------
template <bool EPI>
__global__ __launch_bounds__(256) void wmma_gemm(const _Float16* __restrict__ A,
                                                 const _Float16* __restrict__ BT,
                                                 float* __restrict__ Cm,
                                                 int M, int N, int K, float scale,
                                                 const float* __restrict__ Dv,
                                                 const float* __restrict__ U) {
  const int lane  = threadIdx.x & 31;
  const int wave  = threadIdx.x >> 5;
  const int waveM = wave >> 1;                       // 0..3
  const int waveN = wave & 1;                        // 0..1
  const int bm    = blockIdx.x * 128 + waveM * 32;   // wave M base
  const int bn    = blockIdx.y * 128 + waveN * 64;   // wave N base
  const int lrow  = lane & 15;
  const int khalf = lane >> 4;

  v8f acc[2][4] = {};

  for (int k0 = 0; k0 < K; k0 += 32) {
    V16 a[2], b[4];
#pragma unroll
    for (int mi = 0; mi < 2; ++mi) {
      // A frag: lane M=lrow; halves 0..7 -> K=k0+8*khalf+0..7, 8..15 -> +16
      const _Float16* ap = A + (size_t)(bm + mi * 16 + lrow) * K + k0 + khalf * 8;
      a[mi].h[0] = *(const v8h*)(ap);
      a[mi].h[1] = *(const v8h*)(ap + 16);
    }
#pragma unroll
    for (int ni = 0; ni < 4; ++ni) {
      // B frag: lane N=lrow; halves 0..7 -> K=k0+8*khalf+0..7, 8..15 -> +16
      const _Float16* bp = BT + (size_t)(bn + ni * 16 + lrow) * K + k0 + khalf * 8;
      b[ni].h[0] = *(const v8h*)(bp);
      b[ni].h[1] = *(const v8h*)(bp + 16);
    }
#pragma unroll
    for (int mi = 0; mi < 2; ++mi)
#pragma unroll
      for (int ni = 0; ni < 4; ++ni)
        acc[mi][ni] = __builtin_amdgcn_wmma_f32_16x16x32_f16(
            false, a[mi].v, false, b[ni].v, (short)0, acc[mi][ni], false, false);
  }

#pragma unroll
  for (int mi = 0; mi < 2; ++mi) {
#pragma unroll
    for (int ni = 0; ni < 4; ++ni) {
      const int n = bn + ni * 16 + lrow;
#pragma unroll
      for (int r = 0; r < 8; ++r) {
        const int m = bm + mi * 16 + r + khalf * 8;
        float v = acc[mi][ni][r] * scale;
        if (EPI) v += Dv[n] * U[(size_t)m * N + n];
        Cm[(size_t)m * N + n] = v;
      }
    }
  }
}

// ---------------------------------------------------------------------------
// K5a: in-place local complex scan per (chunk g, channel p). Coalesced over p.
// ---------------------------------------------------------------------------
__global__ __launch_bounds__(256) void scan_local(float* __restrict__ Bu,
                                                  const float* __restrict__ lamBar,
                                                  float* __restrict__ agg) {
  int p = threadIdx.x;                 // 0..255
  int g = blockIdx.x;                  // 0..NCHUNK-1
  float lr = lamBar[2 * p], li = lamBar[2 * p + 1];
  float xr = 0.0f, xi = 0.0f;
  int base = g * CHUNK;
  for (int t = 0; t < CHUNK; ++t) {
    float* row = Bu + (size_t)(base + t) * NS;
    float br = row[p], bi = row[PP + p];
    float nr = fmaf(xr, lr, fmaf(-xi, li, br));
    float ni = fmaf(xr, li, fmaf(xi, lr, bi));
    xr = nr; xi = ni;
    row[p] = xr; row[PP + p] = xi;
  }
  agg[(g * PP + p) * 2]     = xr;
  agg[(g * PP + p) * 2 + 1] = xi;
}

// ---------------------------------------------------------------------------
// K5b: inter-chunk exclusive scan (1 block; A_g = lambda^CHUNK is constant)
// ---------------------------------------------------------------------------
__global__ void scan_chunks(const float* __restrict__ agg,
                            const float* __restrict__ lamC,
                            float* __restrict__ carryIn) {
  int p = threadIdx.x;
  float ar = lamC[2 * p], ai = lamC[2 * p + 1];
  float cr = 0.0f, ci = 0.0f;
  for (int g = 0; g < NCHUNK; ++g) {
    carryIn[(g * PP + p) * 2]     = cr;
    carryIn[(g * PP + p) * 2 + 1] = ci;
    float br = agg[(g * PP + p) * 2], bi = agg[(g * PP + p) * 2 + 1];
    float nr = fmaf(cr, ar, fmaf(-ci, ai, br));
    float ni = fmaf(cr, ai, fmaf(ci, ar, bi));
    cr = nr; ci = ni;
  }
}

// ---------------------------------------------------------------------------
// K5c: fixup x = x_local + lambda^(tl+1) * carry, pack to f16 X(L,2P)
// ---------------------------------------------------------------------------
__global__ __launch_bounds__(256) void scan_fix(const float* __restrict__ Bu,
                                                const float* __restrict__ carryIn,
                                                const float* __restrict__ lampow,
                                                _Float16* __restrict__ X) {
  int q  = blockIdx.x * 256 + threadIdx.x;   // LL*PP threads
  int p  = q & (PP - 1);
  int t  = q >> 8;
  int g  = t >> 7;                           // t / CHUNK
  int tl = t & (CHUNK - 1);
  float cr = carryIn[(g * PP + p) * 2], ci = carryIn[(g * PP + p) * 2 + 1];
  float wr = lampow[(p * CHUNK + tl) * 2], wi = lampow[(p * CHUNK + tl) * 2 + 1];
  float xr = Bu[(size_t)t * NS + p]      + wr * cr - wi * ci;
  float xi = Bu[(size_t)t * NS + PP + p] + wr * ci + wi * cr;
  X[(size_t)t * NS + p]      = (_Float16)xr;
  X[(size_t)t * NS + PP + p] = (_Float16)xi;
}

// ---------------------------------------------------------------------------
extern "C" void kernel_launch(void* const* d_in, const int* in_sizes, int n_in,
                              void* d_out, int out_size, void* d_ws, size_t ws_size,
                              hipStream_t stream) {
  const float* U   = (const float*)d_in[0];  // (L,H)
  const float* Lre = (const float*)d_in[1];  // (P,)
  const float* Lim = (const float*)d_in[2];  // (P,)
  const float* Bm  = (const float*)d_in[3];  // (P,H,2)
  const float* Cmx = (const float*)d_in[4];  // (H,P,2)
  const float* Dv  = (const float*)d_in[5];  // (H,)
  const float* lst = (const float*)d_in[6];  // (P,)
  float* out = (float*)d_out;                // (L,H)

  char* ws = (char*)d_ws;
  size_t off = 0;
  auto alloc = [&](size_t bytes) {
    void* p = ws + off;
    off = (off + bytes + 255) & ~(size_t)255;
    return p;
  };
  float*    lamBar  = (float*)alloc(PP * 2 * sizeof(float));
  float*    coef    = (float*)alloc(PP * 2 * sizeof(float));
  float*    lamC    = (float*)alloc(PP * 2 * sizeof(float));
  float*    lampow  = (float*)alloc((size_t)PP * CHUNK * 2 * sizeof(float));
  _Float16* W1T     = (_Float16*)alloc((size_t)NS * HH * sizeof(_Float16));
  _Float16* V2T     = (_Float16*)alloc((size_t)HH * NS * sizeof(_Float16));
  _Float16* U16     = (_Float16*)alloc((size_t)LL * HH * sizeof(_Float16));
  float*    Bu      = (float*)alloc((size_t)LL * NS * sizeof(float));
  _Float16* X       = (_Float16*)alloc((size_t)LL * NS * sizeof(_Float16));
  float*    agg     = (float*)alloc((size_t)NCHUNK * PP * 2 * sizeof(float));
  float*    carryIn = (float*)alloc((size_t)NCHUNK * PP * 2 * sizeof(float));

  setup_k<<<1, 256, 0, stream>>>(Lre, Lim, lst, lamBar, coef, lamC, lampow);
  build_W1<<<(PP * HH) / 256, 256, 0, stream>>>(Bm, coef, W1T);
  build_V2<<<(HH * PP) / 256, 256, 0, stream>>>(Cmx, V2T);
  cvt_u<<<(LL * HH) / (256 * 4), 256, 0, stream>>>(U, U16);

  // GEMM1: Bu(L,2P) = u @ W1T^T, undo BSCALE
  wmma_gemm<false><<<dim3(LL / 128, NS / 128), 256, 0, stream>>>(
      U16, W1T, Bu, LL, NS, HH, 1.0f / BSCALE, nullptr, nullptr);

  scan_local<<<NCHUNK, 256, 0, stream>>>(Bu, lamBar, agg);
  scan_chunks<<<1, 256, 0, stream>>>(agg, lamC, carryIn);
  scan_fix<<<(LL * PP) / 256, 256, 0, stream>>>(Bu, carryIn, lampow, X);

  // GEMM2: out(L,H) = X @ V2T^T + D*u
  wmma_gemm<true><<<dim3(LL / 128, HH / 128), 256, 0, stream>>>(
      X, V2T, out, LL, HH, NS, 1.0f, Dv, U);
}
